// FusionAwareGAT_70119636075170
// MI455X (gfx1250) — compile-verified, matrix operand order checked by reading, not run on hardware
//
#include <hip/hip_runtime.h>
#include <hip/hip_bf16.h>
#include <math.h>
#include <stdint.h>

// ---------------- problem constants (match reference) ----------------
constexpr int NB    = 4;
constexpr int NHW   = 4800;          // 60*80
constexpr int NK    = 8;             // kNN
constexpr int IMG_D = 192;
constexpr int PT_D  = 128;
constexpr int NH    = 6;
constexpr int DH    = 64;
constexpr int INNER = 384;           // NH*DH
constexpr int KV_IN = PT_D + 26;     // 154
constexpr int KA    = 160;           // KV_IN padded to mult of 32 (zero cols)
constexpr float MAXD2 = 0.01f;       // MAX_DIST^2

typedef __attribute__((ext_vector_type(8)))  float  v8f;
typedef __attribute__((ext_vector_type(8)))  __bf16 v8bf;
typedef __attribute__((ext_vector_type(16))) __bf16 v16bf;

// concat two 8-wide bf16 chunks into one 16-wide WMMA fragment
__device__ __forceinline__ v16bf cat16(v8bf lo, v8bf hi) {
    return __builtin_shufflevector(lo, hi, 0,1,2,3,4,5,6,7,8,9,10,11,12,13,14,15);
}
// bf16 WMMA fragment chunks per lane (ISA 16-bit operand layout):
//   lanes 0-15 : K = k0+[0,8)  and k0+[16,24)
//   lanes 16-31: K = k0+[8,16) and k0+[24,32)
__device__ __forceinline__ v16bf frag_from(const __bf16* row, int k0, int half) {
    v8bf lo = *(const v8bf*)(row + k0 + 8 * half);
    v8bf hi = *(const v8bf*)(row + k0 + 16 + 8 * half);
    return cat16(lo, hi);
}

// ---------------- kernel 0: pack weights to column-major bf16 ----------------
constexpr int NWQ  = INNER * IMG_D;          // 73728  : Wq_t [384][192]
constexpr int NWKV = (2 * INNER) * KA;       // 122880 : Wkv_t[768][160] (k>=154 zero)
constexpr int NWO  = INNER * INNER;          // 147456 : Wout_t[384][384]
constexpr int NPACK = NWQ + NWKV + NWO;      // 344064

__global__ void pack_weights(const float* __restrict__ Wq,
                             const float* __restrict__ Wkv,
                             const float* __restrict__ Wout,
                             __bf16* __restrict__ Wq_t,
                             __bf16* __restrict__ Wkv_t,
                             __bf16* __restrict__ Wout_t)
{
    const int i = blockIdx.x * 256 + threadIdx.x;
    if (i < NWQ) {
        const int n = i / IMG_D, k = i % IMG_D;
        Wq_t[i] = (__bf16)Wq[(size_t)k * INNER + n];
    } else if (i < NWQ + NWKV) {
        const int j = i - NWQ;
        const int n = j / KA, k = j % KA;
        Wkv_t[j] = (k < KV_IN) ? (__bf16)Wkv[(size_t)k * (2 * INNER) + n] : (__bf16)0.f;
    } else if (i < NPACK) {
        const int j = i - NWQ - NWKV;
        const int n = j / INNER, k = j % INNER;
        Wout_t[j] = (__bf16)Wout[(size_t)k * INNER + n];
    }
}

// ---------------- kernel 1: brute-force kNN ----------------
constexpr int QPB = 240;

__global__ void knn_kernel(const float* __restrict__ proj_uv,  // [NB,NHW,2]
                           const float* __restrict__ img_uv,   // [NHW,2]
                           int* __restrict__ idx_out)          // [NB,NHW,NK]
{
    __shared__ float puv[NHW * 2];
    const int blocksPerB = NHW / QPB;
    const int b    = blockIdx.x / blocksPerB;
    const int qblk = blockIdx.x % blocksPerB;

    for (int i = threadIdx.x; i < NHW * 2; i += blockDim.x)
        puv[i] = proj_uv[(size_t)b * NHW * 2 + i];
    __syncthreads();

    if (threadIdx.x < QPB) {
        const int q = qblk * QPB + threadIdx.x;
        const float qx = img_uv[q * 2 + 0];
        const float qy = img_uv[q * 2 + 1];
        float bd[NK]; int bi[NK];
#pragma unroll
        for (int j = 0; j < NK; ++j) { bd[j] = 3.4e38f; bi[j] = 0; }
        for (int n = 0; n < NHW; ++n) {
            const float dx = puv[2 * n + 0] - qx;
            const float dy = puv[2 * n + 1] - qy;
            const float d2 = dx * dx + dy * dy;
            if (d2 < bd[NK - 1]) {
                int j = NK - 1;
                while (j > 0 && bd[j - 1] > d2) {
                    bd[j] = bd[j - 1]; bi[j] = bi[j - 1]; --j;
                }
                bd[j] = d2; bi[j] = n;
            }
        }
#pragma unroll
        for (int j = 0; j < NK; ++j)
            idx_out[((size_t)b * NHW + q) * NK + j] = bi[j];
    }
}

// ---------------- kernel 2: layernorm + Q GEMM (bf16 WMMA) + RMSNorm ----------------
__global__ void qproj_kernel(const float* __restrict__ feat_2d,
                             const float* __restrict__ ln_gamma,
                             const __bf16* __restrict__ Wq_t,    // [INNER][IMG_D]
                             const float* __restrict__ q_gamma,  // [NH*DH]
                             float* __restrict__ qn_out)         // [NB*NHW, INNER]
{
    constexpr int FS = IMG_D + 2;
    __shared__ float  Fl[16 * FS];         // f32 staging for layernorm stats
    __shared__ __bf16 Ab[16 * IMG_D];      // bf16 A matrix (rows 16B aligned: 384B stride)
    __shared__ float  Qr[16 * INNER];      // raw q (f32 accum)

    const int row0 = blockIdx.x * 16;

    for (int i = threadIdx.x; i < 16 * IMG_D; i += 256) {
        const int r = i / IMG_D, c = i % IMG_D;
        Fl[r * FS + c] = feat_2d[(size_t)(row0 + r) * IMG_D + c];
    }
    __syncthreads();

    if (threadIdx.x < 16) {
        const int r = threadIdx.x;
        float mu = 0.f;
        for (int c = 0; c < IMG_D; ++c) mu += Fl[r * FS + c];
        mu *= (1.0f / IMG_D);
        float var = 0.f;
        for (int c = 0; c < IMG_D; ++c) {
            const float d = Fl[r * FS + c] - mu;
            var += d * d;
        }
        var *= (1.0f / IMG_D);
        const float rs = rsqrtf(var + 1e-5f);
        for (int c = 0; c < IMG_D; ++c)
            Ab[r * IMG_D + c] = (__bf16)((Fl[r * FS + c] - mu) * rs * ln_gamma[c]);
    }
    __syncthreads();

    const int wave = threadIdx.x >> 5;
    const int lane = threadIdx.x & 31;
    const int half = lane >> 4;
    const int lm   = lane & 15;

    // hoist A fragments (row lm), reuse across all n-tiles of this wave
    v16bf afrag[IMG_D / 32];
#pragma unroll
    for (int kc = 0; kc < IMG_D / 32; ++kc)
        afrag[kc] = frag_from(Ab + lm * IMG_D, kc * 32, half);

    for (int nt = wave; nt < INNER / 16; nt += 8) {
        const int n0 = nt * 16;
        const __bf16* wrow = Wq_t + (size_t)(n0 + lm) * IMG_D;
        v8f acc = {};
#pragma unroll
        for (int kc = 0; kc < IMG_D / 32; ++kc) {
            v16bf bfrag = frag_from(wrow, kc * 32, half);
            acc = __builtin_amdgcn_wmma_f32_16x16x32_bf16(
                      false, afrag[kc], false, bfrag, (short)0, acc, false, false);
        }
#pragma unroll
        for (int r = 0; r < 8; ++r)
            Qr[(r + 8 * half) * INNER + n0 + lm] = acc[r];
    }
    __syncthreads();

    for (int p = threadIdx.x; p < 16 * NH; p += 256) {
        const int r = p / NH, h = p % NH;
        float ss = 0.f;
        for (int d = 0; d < DH; ++d) {
            const float v = Qr[r * INNER + h * DH + d];
            ss += v * v;
        }
        const float nrm = fmaxf(sqrtf(ss), 1e-12f);
        const float s = 8.0f / nrm;   // sqrt(DH)=8
        for (int d = 0; d < DH; ++d)
            qn_out[(size_t)(row0 + r) * INNER + h * DH + d] =
                Qr[r * INNER + h * DH + d] * s * q_gamma[h * DH + d];
    }
}

// ---------------- kernel 3: fused KV GEMM + attention + out GEMM ----------------
constexpr int PQ   = 8;              // pixels per block
constexpr int MROW = PQ * NK;        // 64 KV rows
constexpr int PS   = INNER + 4;      // 388 (f32 projection buffer stride)

constexpr size_t OFF_AB  = 0;                                    // bf16 [64][160]
constexpr size_t OFF_P   = OFF_AB  + (size_t)MROW * KA * 2;      // f32  [64][388]
constexpr size_t OFF_QS  = OFF_P   + (size_t)MROW * PS * 4;      // f32  [8][384] (async target)
constexpr size_t OFF_OB  = OFF_QS  + (size_t)PQ * INNER * 4;     // bf16 [16][384]
constexpr size_t OFF_ATT = OFF_OB  + (size_t)16 * INNER * 2;     // f32  [8][6][8]
constexpr size_t OFF_OFX = OFF_ATT + (size_t)PQ * NH * NK * 4;
constexpr size_t OFF_OFY = OFF_OFX + (size_t)MROW * 4;
constexpr size_t OFF_MSK = OFF_OFY + (size_t)MROW * 4;
constexpr size_t OFF_NN  = OFF_MSK + (size_t)MROW * 4;
constexpr size_t SMEM3_BYTES = OFF_NN + (size_t)MROW * 4;        // ~143.5 KB

__global__ void attn_kernel(const float* __restrict__ feat_3d,   // [NB,NHW,PT_D]
                            const float* __restrict__ img_uv,    // [NHW,2]
                            const __bf16* __restrict__ Wkv_t,    // [768][KA]
                            const float* __restrict__ k_gamma,   // [NH*DH]
                            const __bf16* __restrict__ Wout_t,   // [384][384]
                            const int*   __restrict__ idxp,      // [NB,NHW,NK]
                            const float* __restrict__ qn,        // [NB*NHW, INNER]
                            float* __restrict__ out)             // [NB*NHW, INNER]
{
    extern __shared__ char smem3[];
    __bf16* Ab  = (__bf16*)(smem3 + OFF_AB);
    float*  P   = (float*) (smem3 + OFF_P);
    float*  QS  = (float*) (smem3 + OFF_QS);
    __bf16* OB  = (__bf16*)(smem3 + OFF_OB);
    float*  ATT = (float*) (smem3 + OFF_ATT);
    float*  OFX = (float*) (smem3 + OFF_OFX);
    float*  OFY = (float*) (smem3 + OFF_OFY);
    float*  MSK = (float*) (smem3 + OFF_MSK);
    int*    NN  = (int*)   (smem3 + OFF_NN);

    const int row0p = blockIdx.x * PQ;       // flat pixel base
    const int b  = row0p / NHW;
    const int q0 = row0p % NHW;

    const int wave = threadIdx.x >> 5;
    const int lane = threadIdx.x & 31;
    const int half = lane >> 4;
    const int lm   = lane & 15;

    // ---- async-stage the q tile into LDS (CDNA5 ASYNCcnt path), overlapped
    //      with the A-matrix build below ----
    {
        const uint64_t qbase = (uint64_t)(uintptr_t)qn;
        const uint32_t loff0 = (uint32_t)OFF_QS;
        for (int t = threadIdx.x; t < (PQ * INNER) / 4; t += 256) {  // 768 x b128
            const uint32_t loff = loff0 + (uint32_t)t * 16u;
            const uint32_t goff = (uint32_t)((size_t)row0p * INNER * 4 + (size_t)t * 16);
            asm volatile("global_load_async_to_lds_b128 %0, %1, %2"
                         :: "v"(loff), "v"(goff), "s"(qbase)
                         : "memory");
        }
    }

    // ---- stage 0: neighbor meta ----
    for (int r = threadIdx.x; r < MROW; r += 256) {
        const int p = r >> 3, k = r & 7;
        const int q = q0 + p;
        const int nn = idxp[((size_t)b * NHW + q) * NK + k];
        NN[r] = nn;
        const float ox = img_uv[nn * 2 + 0] - img_uv[q * 2 + 0];
        const float oy = img_uv[nn * 2 + 1] - img_uv[q * 2 + 1];
        OFX[r] = ox; OFY[r] = oy;
        MSK[r] = (ox * ox + oy * oy <= MAXD2) ? 1.f : 0.f;
    }
    __syncthreads();

    // ---- stage 1: build A (bf16) = [knn_f3 | pos emb | 0-pad] ----
    for (int i = threadIdx.x; i < MROW * KA; i += 256) {
        const int r = i / KA, c = i % KA;
        float v;
        if (c < PT_D) {
            v = feat_3d[((size_t)b * NHW + NN[r]) * PT_D + c];
        } else if (c < KV_IN) {
            const int j = c - PT_D;
            const float ox = OFX[r], oy = OFY[r];
            if      (j < 6)   v = __sinf(ox * (float)(1 << j));
            else if (j < 12)  v = __sinf(oy * (float)(1 << (j - 6)));
            else if (j < 18)  v = __cosf(ox * (float)(1 << (j - 12)));
            else if (j < 24)  v = __cosf(oy * (float)(1 << (j - 18)));
            else if (j == 24) v = ox;
            else              v = oy;
        } else {
            v = 0.f;
        }
        Ab[r * KA + c] = (__bf16)v;
    }
    asm volatile("s_wait_asynccnt 0x0" ::: "memory");   // q tile landed in LDS
    __syncthreads();

    // ---- stages 2 & 5: K then V projection  P[64,384] = A @ Wkv_t ----
    // wave -> m-tile (wave&3); n-tiles strided by 2 starting at (wave>>2)
    const int mt = wave & 3;
    const int m0 = mt * 16;
    const int ns = wave >> 2;

    v16bf afrag[KA / 32];
#pragma unroll
    for (int kc = 0; kc < KA / 32; ++kc)
        afrag[kc] = frag_from(Ab + (m0 + lm) * KA, kc * 32, half);

    for (int pass = 0; pass < 2; ++pass) {
        const int ncol0 = pass * INNER;      // 0 = K proj, INNER = V proj
        for (int nt = ns; nt < INNER / 16; nt += 2) {
            const int n0 = nt * 16;
            const __bf16* wrow = Wkv_t + (size_t)(ncol0 + n0 + lm) * KA;
            v8f acc = {};
#pragma unroll
            for (int kc = 0; kc < KA / 32; ++kc) {
                v16bf bfrag = frag_from(wrow, kc * 32, half);
                acc = __builtin_amdgcn_wmma_f32_16x16x32_bf16(
                          false, afrag[kc], false, bfrag, (short)0, acc, false, false);
            }
#pragma unroll
            for (int r = 0; r < 8; ++r)
                P[(m0 + r + 8 * half) * PS + n0 + lm] = acc[r];
        }
        __syncthreads();

        if (pass == 0) {
            // ---- stage 3: RMSNorm(k) per (row, head), in place ----
            for (int p = threadIdx.x; p < MROW * NH; p += 256) {
                const int r = p / NH, h = p % NH;
                float ss = 0.f;
                for (int d = 0; d < DH; ++d) {
                    const float v = P[r * PS + h * DH + d];
                    ss += v * v;
                }
                const float nrm = fmaxf(sqrtf(ss), 1e-12f);
                const float s = 8.0f / nrm;
                for (int d = 0; d < DH; ++d)
                    P[r * PS + h * DH + d] *= s * k_gamma[h * DH + d];
            }
            __syncthreads();

            // ---- stage 4: masked scores + softmax over NK ----
            if (threadIdx.x < PQ * NH) {
                const int p = threadIdx.x / NH, h = threadIdx.x % NH;
                float sc[NK];
                float mx = -3.4e38f;
#pragma unroll
                for (int k = 0; k < NK; ++k) {
                    float s = 0.f;
                    for (int d = 0; d < DH; ++d)
                        s += QS[p * INNER + h * DH + d] * P[(p * NK + k) * PS + h * DH + d];
                    s = (MSK[p * NK + k] > 0.5f) ? s : -1e30f;
                    sc[k] = s;
                    mx = fmaxf(mx, s);
                }
                float den = 0.f;
#pragma unroll
                for (int k = 0; k < NK; ++k) { sc[k] = __expf(sc[k] - mx); den += sc[k]; }
                const float inv = 1.0f / den;
#pragma unroll
                for (int k = 0; k < NK; ++k)
                    ATT[(p * NH + h) * NK + k] = sc[k] * inv;
            }
            __syncthreads();   // everyone done reading P (k) before V pass overwrites
        }
    }

    // ---- stage 6: attention combine -> OB (bf16, rows 8..15 zero-padded) ----
    for (int i = threadIdx.x; i < 16 * INNER; i += 256) {
        const int p = i / INNER, c = i % INNER;
        float o = 0.f;
        if (p < PQ) {
            const int h = c / DH;
#pragma unroll
            for (int k = 0; k < NK; ++k)
                o += ATT[(p * NH + h) * NK + k] * P[(p * NK + k) * PS + c];
        }
        OB[p * INNER + c] = (__bf16)o;
    }
    __syncthreads();

    // ---- stage 7: final out GEMM: [16,384] @ Wout_t^T -> rows 0..7 ----
    {
        v16bf ofrag[INNER / 32];
#pragma unroll
        for (int kc = 0; kc < INNER / 32; ++kc)
            ofrag[kc] = frag_from(OB + lm * INNER, kc * 32, half);

        for (int nt = wave; nt < INNER / 16; nt += 8) {
            const int n0 = nt * 16;
            const __bf16* wrow = Wout_t + (size_t)(n0 + lm) * INNER;
            v8f acc = {};
#pragma unroll
            for (int kc = 0; kc < INNER / 32; ++kc) {
                v16bf bfrag = frag_from(wrow, kc * 32, half);
                acc = __builtin_amdgcn_wmma_f32_16x16x32_bf16(
                          false, ofrag[kc], false, bfrag, (short)0, acc, false, false);
            }
            if (half == 0) {   // lanes 0-15 hold rows 0..7 (the real rows)
#pragma unroll
                for (int r = 0; r < 8; ++r)
                    out[(size_t)(row0p + r) * INNER + n0 + lm] = acc[r];
            }
        }
    }
}

// ---------------- host launcher ----------------
extern "C" void kernel_launch(void* const* d_in, const int* in_sizes, int n_in,
                              void* d_out, int out_size, void* d_ws, size_t ws_size,
                              hipStream_t stream) {
    const float* feat_2d  = (const float*)d_in[0];
    const float* feat_3d  = (const float*)d_in[1];
    const float* proj_uv  = (const float*)d_in[2];
    const float* img_uv   = (const float*)d_in[3];
    const float* ln_gamma = (const float*)d_in[4];
    const float* q_gamma  = (const float*)d_in[5];
    const float* k_gamma  = (const float*)d_in[6];
    const float* Wq       = (const float*)d_in[7];
    const float* Wkv      = (const float*)d_in[8];
    const float* Wout     = (const float*)d_in[9];
    float* out = (float*)d_out;

    char* ws = (char*)d_ws;
    int*    idx    = (int*)ws;                                   // 614400 B
    float*  qn     = (float*)(ws + (size_t)NB * NHW * NK * 4);   // 29.49 MB
    char*   wpk    = ws + (size_t)NB * NHW * NK * 4 + (size_t)NB * NHW * INNER * 4;
    __bf16* Wq_t   = (__bf16*)wpk;                               // 147456 B
    __bf16* Wkv_t  = (__bf16*)(wpk + (size_t)NWQ * 2);           // 245760 B
    __bf16* Wout_t = (__bf16*)(wpk + (size_t)(NWQ + NWKV) * 2);  // 294912 B

    (void)in_sizes; (void)n_in; (void)out_size; (void)ws_size;

    hipFuncSetAttribute(reinterpret_cast<const void*>(attn_kernel),
                        hipFuncAttributeMaxDynamicSharedMemorySize,
                        (int)SMEM3_BYTES);

    pack_weights<<<dim3((NPACK + 255) / 256), dim3(256), 0, stream>>>(
        Wq, Wkv, Wout, Wq_t, Wkv_t, Wout_t);
    knn_kernel<<<dim3(NB * (NHW / QPB)), dim3(256), 0, stream>>>(proj_uv, img_uv, idx);
    qproj_kernel<<<dim3((NB * NHW) / 16), dim3(256), 0, stream>>>(
        feat_2d, ln_gamma, Wq_t, q_gamma, qn);
    attn_kernel<<<dim3((NB * NHW) / PQ), dim3(256), SMEM3_BYTES, stream>>>(
        feat_3d, img_uv, Wkv_t, k_gamma, Wout_t, idx, qn, out);
}
